// TransformerEncoderLevel_15436112462322
// MI455X (gfx1250) — matrix operand order repaired and checked by reading, not compiled, not run on hardware
//
#include <hip/hip_runtime.h>
#include <hip/hip_bf16.h>
#include <math.h>

// ---------------------------------------------------------------------------
// Transformer encoder layer for MI455X (gfx1250), bf16 WMMA everywhere.
// B=2, S=2048, D=1024, H=16, dh=64, DA=1024, FFN=4096, mask_size=128.
// GEMMs: 128x128 block tile, 4 waves (64x64 wave tile = 16 WMMA / K-step),
// double-buffered LDS, weights pre-transposed so all LDS fragment loads are
// contiguous ds_load_b128. Async global->LDS copies (ASYNCcnt) when the
// toolchain exposes the gfx1250 builtin.
// ---------------------------------------------------------------------------

typedef __bf16 bf16_t;
typedef __attribute__((ext_vector_type(16))) __bf16 v16bf;
typedef __attribute__((ext_vector_type(8)))  float  v8f;

#define WMMA_BF16(a, b, c)                                                     \
  __builtin_amdgcn_wmma_f32_16x16x32_bf16(false, (a), false, (b), (short)0,    \
                                          (c), false, false)

#if defined(__HIP_DEVICE_COMPILE__) &&                                         \
    __has_builtin(__builtin_amdgcn_global_load_async_to_lds_b128)
#define ASYNC_LDS 1
#endif

static constexpr int Bb   = 2;
static constexpr int Ss   = 2048;
static constexpr int Dd   = 1024;
static constexpr int Hh   = 16;
static constexpr int DAa  = 1024;
static constexpr int FFNn = 4096;
static constexpr int MR   = Bb * Ss;  // 4096 rows
static constexpr int MSK  = 128;      // mask_size (block-diagonal window)

// ---------------------------------------------------------------------------
// 16-byte global -> LDS copy; async (ASYNCcnt-tracked) when available.
// Builtin signature (from hipcc diagnostic): param 0 is
// 'int __attribute__((vector_size(16))) __device__ *' i.e. v4i addrspace(1)*.
typedef int v4i_t __attribute__((vector_size(16)));
typedef __attribute__((address_space(1))) v4i_t* gvec_p;
typedef __attribute__((address_space(3))) v4i_t* lvec_p;

__device__ __forceinline__ void copy16_g2l(const bf16_t* __restrict__ g,
                                           bf16_t* __restrict__ l) {
#ifdef ASYNC_LDS
  __builtin_amdgcn_global_load_async_to_lds_b128((gvec_p)g, (lvec_p)l, 0, 0);
#else
  *reinterpret_cast<uint4*>(l) = *reinterpret_cast<const uint4*>(g);
#endif
}
__device__ __forceinline__ void copy_wait() {
#ifdef ASYNC_LDS
  asm volatile("s_wait_asynccnt 0x0" ::: "memory");
#endif
}

// ---------------------------------------------------------------------------
// WMMA fragment loaders (wave32).  See CDNA5 ISA §7.12.2.
// A 16x32 bf16 row-major: lanes 0-15 -> M=lane, K=0..15; lanes 16-31 ->
// M=lane-16, K=16..31 (matches B fragments).  Two 16B reads/lane.
__device__ __forceinline__ v16bf frag_a(const bf16_t* __restrict__ src, int ld) {
  const int lane = threadIdx.x & 31;
  const int kb = (lane >> 4) * 16;
  const bf16_t* row = src + (size_t)(lane & 15) * ld;
  v16bf f;
#pragma unroll
  for (int j = 0; j < 4; ++j) {
    f[2 * j]     = row[kb + 2 * j];
    f[2 * j + 1] = row[kb + 2 * j + 1];
  }
#pragma unroll
  for (int j = 0; j < 4; ++j) {
    f[8 + 2 * j]     = row[8 + kb + 2 * j];
    f[8 + 2 * j + 1] = row[8 + kb + 2 * j + 1];
  }
  return f;
}

// B 32x16 bf16 where B^T is stored ROW-MAJOR [N x K] with leading dim ld:
// lane n = lane&15 holds column n; lanes 0-15 K=0..15, lanes 16-31 K=16..31.
// One contiguous 32B read per lane.
__device__ __forceinline__ v16bf frag_bn(const bf16_t* __restrict__ src, int ld) {
  const int lane = threadIdx.x & 31;
  const int kb = (lane >> 4) * 16;
  const bf16_t* row = src + (size_t)(lane & 15) * ld;
  v16bf f;
#pragma unroll
  for (int j = 0; j < 8; ++j) {
    f[2 * j]     = row[kb + 2 * j];
    f[2 * j + 1] = row[kb + 2 * j + 1];
  }
  return f;
}

// B 32x16 bf16, B stored ROW-MAJOR [K x N] (strided; only used in the small
// attention P*V step).
__device__ __forceinline__ v16bf frag_bk(const bf16_t* __restrict__ src, int ld) {
  const int lane = threadIdx.x & 31;
  const int n  = lane & 15;
  const int kb = (lane >> 4) * 16;
  v16bf f;
#pragma unroll
  for (int j = 0; j < 8; ++j) {
    f[2 * j]     = src[(size_t)(kb + 2 * j) * ld + n];
    f[2 * j + 1] = src[(size_t)(kb + 2 * j + 1) * ld + n];
  }
  return f;
}

__device__ __forceinline__ float gelu_exact(float x) {
  return 0.5f * x * (1.0f + erff(x * 0.70710678118654752f));
}

// ---------------------------------------------------------------------------
// fp32 -> bf16 conversion (activations)
__global__ __launch_bounds__(256) void cvt_bf16_kernel(
    const float* __restrict__ in, bf16_t* __restrict__ out, int n) {
  const int i = blockIdx.x * 256 + threadIdx.x;
  if (i < n) out[i] = (bf16_t)in[i];
}

// fp32 [K x N] -> bf16 transposed [N x K] (weights), 32x32 LDS tile.
__global__ __launch_bounds__(256) void cvt_tr_bf16_kernel(
    const float* __restrict__ in, bf16_t* __restrict__ out, int Kd, int Nd) {
  __shared__ bf16_t tile[32][33];
  const int bn = blockIdx.x * 32;  // n
  const int bk = blockIdx.y * 32;  // k
  const int tx = threadIdx.x & 31, ty = threadIdx.x >> 5;  // 32 x 8
#pragma unroll
  for (int i = 0; i < 32; i += 8)
    tile[ty + i][tx] = (bf16_t)in[(size_t)(bk + ty + i) * Nd + bn + tx];
  __syncthreads();
#pragma unroll
  for (int i = 0; i < 32; i += 8)
    out[(size_t)(bn + ty + i) * Kd + bk + tx] = tile[tx][ty + i];
}

// ---------------------------------------------------------------------------
// Tiled bf16 GEMM: C[MxN] = A[MxK] * Bt^T (+bias) (+GELU), Bt is [N x K].
// 128 threads (4 waves, 2x2), wave tile 64x64 -> 16 WMMA per K-step.
// Double-buffered LDS; tile k+1 staged (async when available) during compute.
// ACT: 0 = none, 1 = exact GELU. Writes fp32 (Cf) and/or bf16 (Cb).
template <int ACT>
__global__ __launch_bounds__(128) void gemm_bf16_kernel(
    const bf16_t* __restrict__ A, const bf16_t* __restrict__ Bt,
    const float* __restrict__ bias, float* __restrict__ Cf,
    bf16_t* __restrict__ Cb, int M, int N, int K) {
  __shared__ alignas(16) bf16_t As[2][128 * 32];
  __shared__ alignas(16) bf16_t Bs[2][128 * 32];  // n-major [n][k]
  const int tid  = threadIdx.x;
  const int lane = tid & 31;
  const int wave = tid >> 5;
  const int wm   = (wave & 1) * 64;
  const int wn   = (wave >> 1) * 64;
  const int bm   = blockIdx.x * 128;
  const int bn   = blockIdx.y * 128;

  const v8f z8 = {0.f, 0.f, 0.f, 0.f, 0.f, 0.f, 0.f, 0.f};
  v8f acc[4][4];
#pragma unroll
  for (int i = 0; i < 4; ++i)
#pragma unroll
    for (int j = 0; j < 4; ++j) acc[i][j] = z8;

  // Per-thread staging: thread t copies row t of the A tile (32 bf16) and
  // row t of the Bt tile (32 bf16); both are contiguous 64B in global.
  const bf16_t* gA = A + (size_t)(bm + tid) * K;
  const bf16_t* gB = Bt + (size_t)(bn + tid) * K;

  auto stage = [&](int kt, int buf) {
    const bf16_t* sa = gA + kt * 32;
    const bf16_t* sb = gB + kt * 32;
    bf16_t* da = &As[buf][tid * 32];
    bf16_t* db = &Bs[buf][tid * 32];
#pragma unroll
    for (int c = 0; c < 4; ++c) copy16_g2l(sa + 8 * c, da + 8 * c);
#pragma unroll
    for (int c = 0; c < 4; ++c) copy16_g2l(sb + 8 * c, db + 8 * c);
  };

  const int nk = K >> 5;
  stage(0, 0);
  copy_wait();
  __syncthreads();

  for (int kt = 0; kt < nk; ++kt) {
    const int buf = kt & 1;
    if (kt + 1 < nk) stage(kt + 1, buf ^ 1);
    v16bf af[4], bv[4];
#pragma unroll
    for (int i = 0; i < 4; ++i) af[i] = frag_a(&As[buf][(wm + 16 * i) * 32], 32);
#pragma unroll
    for (int j = 0; j < 4; ++j) bv[j] = frag_bn(&Bs[buf][(wn + 16 * j) * 32], 32);
#pragma unroll
    for (int i = 0; i < 4; ++i)
#pragma unroll
      for (int j = 0; j < 4; ++j) acc[i][j] = WMMA_BF16(af[i], bv[j], acc[i][j]);
    copy_wait();
    __syncthreads();
  }

  const int n_in = lane & 15, mh = lane >> 4;
#pragma unroll
  for (int i = 0; i < 4; ++i) {
#pragma unroll
    for (int j = 0; j < 4; ++j) {
      const int col = bn + wn + 16 * j + n_in;
      const float bvl = bias ? bias[col] : 0.0f;
#pragma unroll
      for (int r = 0; r < 8; ++r) {
        const int row = bm + wm + 16 * i + r + 8 * mh;
        float v = acc[i][j][r] + bvl;
        if (ACT == 1) v = gelu_exact(v);
        if (Cf) Cf[(size_t)row * N + col] = v;
        if (Cb) Cb[(size_t)row * N + col] = (bf16_t)v;
      }
    }
  }
}

// ---------------------------------------------------------------------------
// Fused block-diagonal attention. One workgroup per (query-block, head, batch).
// Q/K/V are bf16 [B*S, DA] row-major (head h occupies cols h*64..h*64+63).
// Output O bf16 [B*S, DA] (== concat of heads, ready for the Wo GEMM).
__global__ __launch_bounds__(256) void attn_kernel(
    const bf16_t* __restrict__ Q, const bf16_t* __restrict__ Km,
    const bf16_t* __restrict__ V, const int* __restrict__ vlens,
    bf16_t* __restrict__ O) {
  __shared__ alignas(16) bf16_t kv[128 * 64];      // K tile, then reused for V
  __shared__ alignas(16) bf16_t probs[128 * 128];  // softmax probabilities
  const int qb = blockIdx.x, h = blockIdx.y, b = blockIdx.z;
  const int tid = threadIdx.x, wave = tid >> 5, lane = tid & 31;
  const int n_in = lane & 15, mh = lane >> 4;
  const int rowbase = b * Ss + qb * MSK;
  const int vlen = vlens[b];

  // Stage K block (row-major [key][d], 64 dims of head h)
  const int skey = tid >> 1, sd0 = (tid & 1) * 32;
  {
    const bf16_t* s = Km + (size_t)(rowbase + skey) * DAa + h * 64 + sd0;
    bf16_t* d = &kv[skey * 64 + sd0];
#pragma unroll
    for (int c = 0; c < 4; ++c) copy16_g2l(s + 8 * c, d + 8 * c);
  }
  copy_wait();
  __syncthreads();

  // Q fragments for this wave's 16 query rows (dh=64 -> two K=32 fragments)
  v16bf aq0, aq1;
  {
    const bf16_t* qsrc = Q + (size_t)(rowbase + wave * 16) * DAa + h * 64;
    aq0 = frag_a(qsrc, DAa);
    aq1 = frag_a(qsrc + 32, DAa);
  }

  // scores = Q * K^T : 8 key tiles of 16, fp32 accumulators
  const v8f z8 = {0.f, 0.f, 0.f, 0.f, 0.f, 0.f, 0.f, 0.f};
  v8f sc[8];
#pragma unroll
  for (int t = 0; t < 8; ++t) sc[t] = z8;
#pragma unroll
  for (int t = 0; t < 8; ++t) {
    sc[t] = WMMA_BF16(aq0, frag_bn(&kv[16 * t * 64], 64), sc[t]);
    sc[t] = WMMA_BF16(aq1, frag_bn(&kv[16 * t * 64 + 32], 64), sc[t]);
  }
  __syncthreads();  // everyone done reading K from kv

  // Stage V block into the same LDS buffer (row-major [key][d]); overlaps
  // with the register softmax below.
  {
    const bf16_t* s = V + (size_t)(rowbase + skey) * DAa + h * 64 + sd0;
    bf16_t* d = &kv[skey * 64 + sd0];
#pragma unroll
    for (int c = 0; c < 4; ++c) copy16_g2l(s + 8 * c, d + 8 * c);
  }

  // Row softmax over the 128 block keys (off-block keys carry exp(NEG)~=0 in
  // the reference's denominator, negligible). C-fragment row r+8*mh lives in a
  // fixed 16-lane half, so shfl_xor masks 1/2/4/8 reduce across key columns.
#pragma unroll
  for (int r = 0; r < 8; ++r) {
    float v[8];
    float mx = -3.0e38f;
#pragma unroll
    for (int t = 0; t < 8; ++t) {
      float sv = sc[t][r] * 0.125f;  // 1/sqrt(dh)
      const int key = qb * MSK + 16 * t + n_in;
      if (key >= vlen) sv = -1.0e9f;  // NEG mask
      v[t] = sv;
      mx = fmaxf(mx, sv);
    }
#pragma unroll
    for (int m = 8; m >= 1; m >>= 1) mx = fmaxf(mx, __shfl_xor(mx, m, 32));
    float sum = 0.f;
#pragma unroll
    for (int t = 0; t < 8; ++t) {
      v[t] = __expf(v[t] - mx);
      sum += v[t];
    }
#pragma unroll
    for (int m = 8; m >= 1; m >>= 1) sum += __shfl_xor(sum, m, 32);
    const float inv = 1.0f / sum;
    const int row = wave * 16 + r + 8 * mh;
#pragma unroll
    for (int t = 0; t < 8; ++t)
      probs[row * 128 + 16 * t + n_in] = (bf16_t)(v[t] * inv);
  }
  copy_wait();
  __syncthreads();

  // out = probs * V : 4 dim tiles of 16, K loop over 128 keys (4 x 32)
  v8f oc[4];
#pragma unroll
  for (int j = 0; j < 4; ++j) oc[j] = z8;
#pragma unroll
  for (int kk = 0; kk < 4; ++kk) {
    v16bf pa = frag_a(&probs[(wave * 16) * 128 + 32 * kk], 128);
#pragma unroll
    for (int j = 0; j < 4; ++j)
      oc[j] = WMMA_BF16(pa, frag_bk(&kv[(32 * kk) * 64 + 16 * j], 64), oc[j]);
  }
#pragma unroll
  for (int j = 0; j < 4; ++j) {
#pragma unroll
    for (int r = 0; r < 8; ++r) {
      const int row = rowbase + wave * 16 + r + 8 * mh;
      O[(size_t)row * DAa + h * 64 + 16 * j + n_in] = (bf16_t)oc[j][r];
    }
  }
}

// ---------------------------------------------------------------------------
// Residual add + LayerNorm over D=1024. One row per block, 256 threads.
// Writes fp32 (Yf) and/or bf16 (Yb).
__global__ __launch_bounds__(256) void addnorm_kernel(
    const float* __restrict__ A, const float* __restrict__ Bm,
    const float* __restrict__ g, const float* __restrict__ bt,
    float* __restrict__ Yf, bf16_t* __restrict__ Yb) {
  const int row = blockIdx.x;
  const int tid = threadIdx.x;
  const size_t base = (size_t)row * Dd;
  float x[4];
  float s = 0.f, s2 = 0.f;
#pragma unroll
  for (int i = 0; i < 4; ++i) {
    const int c = tid + 256 * i;
    const float v = A[base + c] + Bm[base + c];
    x[i] = v;
    s += v;
    s2 += v * v;
  }
#pragma unroll
  for (int m = 16; m >= 1; m >>= 1) {
    s += __shfl_xor(s, m, 32);
    s2 += __shfl_xor(s2, m, 32);
  }
  __shared__ float rs[8], rs2[8];
  if ((tid & 31) == 0) {
    rs[tid >> 5] = s;
    rs2[tid >> 5] = s2;
  }
  __syncthreads();
  float ts = 0.f, t2 = 0.f;
#pragma unroll
  for (int w = 0; w < 8; ++w) {
    ts += rs[w];
    t2 += rs2[w];
  }
  const float mu = ts * (1.0f / Dd);
  const float var = t2 * (1.0f / Dd) - mu * mu;
  const float inv = rsqrtf(var + 1e-5f);
#pragma unroll
  for (int i = 0; i < 4; ++i) {
    const int c = tid + 256 * i;
    const float v = (x[i] - mu) * inv * g[c] + bt[c];
    if (Yf) Yf[base + c] = v;
    if (Yb) Yb[base + c] = (bf16_t)v;
  }
}

// ---------------------------------------------------------------------------
extern "C" void kernel_launch(void* const* d_in, const int* in_sizes, int n_in,
                              void* d_out, int out_size, void* d_ws,
                              size_t ws_size, hipStream_t stream) {
  // Inputs (setup_inputs order)
  const float* X   = (const float*)d_in[0];
  const float* Wq  = (const float*)d_in[1];
  const float* Wk  = (const float*)d_in[2];
  const float* Wv  = (const float*)d_in[3];
  const float* Wo  = (const float*)d_in[4];
  const float* W1  = (const float*)d_in[5];
  const float* b1  = (const float*)d_in[6];
  const float* W2  = (const float*)d_in[7];
  const float* b2  = (const float*)d_in[8];
  const float* g1  = (const float*)d_in[9];
  const float* be1 = (const float*)d_in[10];
  const float* g2  = (const float*)d_in[11];
  const float* be2 = (const float*)d_in[12];
  const int* vlens = (const int*)d_in[13];
  // d_in[14] = mask_size (=128, baked into MSK)

  // Workspace layout (bytes, 256-aligned). Total ~152 MB.
  char* ws = (char*)d_ws;
  size_t off = 0;
  auto alloc = [&](size_t bytes) -> char* {
    char* p = ws + off;
    off += (bytes + 255) & ~(size_t)255;
    return p;
  };
  bf16_t* Xbf  = (bf16_t*)alloc((size_t)MR * Dd * 2);
  bf16_t* Wqt  = (bf16_t*)alloc((size_t)Dd * DAa * 2);   // [DA x D]
  bf16_t* Wkt  = (bf16_t*)alloc((size_t)Dd * DAa * 2);
  bf16_t* Wvt  = (bf16_t*)alloc((size_t)Dd * DAa * 2);
  bf16_t* Wot  = (bf16_t*)alloc((size_t)DAa * Dd * 2);   // [D x DA]
  bf16_t* W1t  = (bf16_t*)alloc((size_t)Dd * FFNn * 2);  // [FFN x D]
  bf16_t* W2t  = (bf16_t*)alloc((size_t)FFNn * Dd * 2);  // [D x FFN]
  bf16_t* Qb   = (bf16_t*)alloc((size_t)MR * DAa * 2);
  bf16_t* Kb   = (bf16_t*)alloc((size_t)MR * DAa * 2);
  bf16_t* Vb   = (bf16_t*)alloc((size_t)MR * DAa * 2);
  bf16_t* Ob   = (bf16_t*)alloc((size_t)MR * DAa * 2);
  float*  Pf   = (float*)alloc((size_t)MR * Dd * 4);
  float*  Yf   = (float*)alloc((size_t)MR * Dd * 4);
  bf16_t* Yb   = (bf16_t*)alloc((size_t)MR * Dd * 2);
  bf16_t* H1b  = (bf16_t*)alloc((size_t)MR * FFNn * 2);
  float*  Ff   = (float*)alloc((size_t)MR * Dd * 4);
  (void)ws_size;
  (void)n_in;
  (void)in_sizes;
  (void)out_size;

  // Activations: straight convert. Weights: convert + transpose to [N x K].
  cvt_bf16_kernel<<<(MR * Dd + 255) / 256, 256, 0, stream>>>(X, Xbf, MR * Dd);
  auto cvtT = [&](const float* src, bf16_t* dst, int Kd, int Nd) {
    dim3 g(Nd / 32, Kd / 32);
    cvt_tr_bf16_kernel<<<g, 256, 0, stream>>>(src, dst, Kd, Nd);
  };
  cvtT(Wq, Wqt, Dd, DAa);
  cvtT(Wk, Wkt, Dd, DAa);
  cvtT(Wv, Wvt, Dd, DAa);
  cvtT(Wo, Wot, DAa, Dd);
  cvtT(W1, W1t, Dd, FFNn);
  cvtT(W2, W2t, FFNn, Dd);

  // Q/K/V projections (bf16 out only)
  dim3 gQKV(MR / 128, DAa / 128);
  gemm_bf16_kernel<0><<<gQKV, 128, 0, stream>>>(Xbf, Wqt, nullptr, nullptr, Qb,
                                                MR, DAa, Dd);
  gemm_bf16_kernel<0><<<gQKV, 128, 0, stream>>>(Xbf, Wkt, nullptr, nullptr, Kb,
                                                MR, DAa, Dd);
  gemm_bf16_kernel<0><<<gQKV, 128, 0, stream>>>(Xbf, Wvt, nullptr, nullptr, Vb,
                                                MR, DAa, Dd);

  // Block-diagonal masked attention
  dim3 gAttn(Ss / MSK, Hh, Bb);
  attn_kernel<<<gAttn, 256, 0, stream>>>(Qb, Kb, Vb, vlens, Ob);

  // Output projection (fp32 out)
  dim3 gWo(MR / 128, Dd / 128);
  gemm_bf16_kernel<0><<<gWo, 128, 0, stream>>>(Ob, Wot, nullptr, Pf, nullptr,
                                               MR, Dd, DAa);

  // AddNorm 1: Y = LN(X + P); fp32 for residual-2, bf16 for FFN GEMM
  addnorm_kernel<<<MR, 256, 0, stream>>>(X, Pf, g1, be1, Yf, Yb);

  // FFN1: H1 = gelu(Y @ W1 + b1), bf16 out
  dim3 gF1(MR / 128, FFNn / 128);
  gemm_bf16_kernel<1><<<gF1, 128, 0, stream>>>(Yb, W1t, b1, nullptr, H1b, MR,
                                               FFNn, Dd);

  // FFN2: F = H1 @ W2 + b2, fp32 out
  dim3 gF2(MR / 128, Dd / 128);
  gemm_bf16_kernel<0><<<gF2, 128, 0, stream>>>(H1b, W2t, b2, Ff, nullptr, MR,
                                               Dd, FFNn);

  // AddNorm 2: Z = LN(Y + F) -> d_out (fp32)
  addnorm_kernel<<<MR, 256, 0, stream>>>(Yf, Ff, g2, be2, (float*)d_out,
                                         nullptr);
}
